// Net_51531017617491
// MI455X (gfx1250) — matrix operand order repaired and checked by reading, not compiled, not run on hardware
//
#include <hip/hip_runtime.h>

// ---------------------------------------------------------------------------
// Fused LSTM (B=8192, T=128, F=32, H=64) + linear head (100 assets) + softmax
// for MI455X / gfx1250. Single fused kernel: hs never touches HBM.
//
// Roofline: ~65 GFLOP vs ~140 MB mandatory HBM traffic (x stream) -> memory/
// latency bound (~6 us floor at 23.3 TB/s). Design: 512 blocks x 4 waves,
// 16 batch rows per block, serial over T with WMMA bf16 (f32 accum).
// v2 changes: weight B-fragments + bias + cell state hoisted to registers
// (loop-invariant / lane-stationary), sX/sWlin double-buffered so the loop
// needs only 2 barriers per timestep, LDS cut to 48 KB (3 blocks/WGP).
// ---------------------------------------------------------------------------

typedef __attribute__((ext_vector_type(16))) __bf16 v16bf;
typedef __attribute__((ext_vector_type(8)))  float  v8f;

#define T_STEPS  128
#define F_IN     32
#define HID      64
#define GATES    256   // 4*HID, gate order i,f,g,o
#define NA       100   // assets
#define NAP      112   // padded to 7 tiles of 16
#define ROWS     16    // batch rows per workgroup (one WMMA M-tile)
#define NTHREADS 128   // 4 waves of 32

// LDS layout (bytes): total 49152 B (48 KB) per workgroup.
#define OFF_WLIN  0                          // bf16 [2][112][64]  28672
#define OFF_H     (OFF_WLIN  + 28672)        // bf16 [16][64]       2048
#define OFF_X     (OFF_H     + 2048)         // bf16 [2][16][32]    2048
#define OFF_GATES (OFF_X     + 2048)         // f32  [16][256]     16384
#define SMEM_BYTES (OFF_GATES + 16384)
// logits [16][112] f32 (7168 B) reuses the sGates region after the loop.

__device__ __forceinline__ float sigmoid_f(float x) {
    return 1.0f / (1.0f + __expf(-x));
}
__device__ __forceinline__ float tanh_f(float x) {
    x = fminf(fmaxf(x, -15.0f), 15.0f);
    float e = __expf(2.0f * x);
    return (e - 1.0f) / (e + 1.0f);
}
__device__ __forceinline__ v8f wmma_bf16(v16bf a, v16bf b, v8f c) {
    // D = A(16x32 bf16) * B(32x16 bf16) + C(16x16 f32)
    return __builtin_amdgcn_wmma_f32_16x16x32_bf16(
        /*neg_a=*/false, a, /*neg_b=*/false, b,
        /*c_mod=*/(short)0, c, /*reuse_a=*/false, /*reuse_b=*/false);
}

// A-fragment (16x32 bf16) element->K map: lanes 0-15 (row M=lane) hold
// K=0..15, lanes 16-31 hold K=16..31 (contiguous, matches B), 2 bf16 per VGPR.
__device__ __forceinline__ int a_kmap(int e, int hi) {
    return hi * 16 + e;
}

__global__ __launch_bounds__(NTHREADS)
void lstm_head_kernel(const float* __restrict__ x,
                      const float* __restrict__ Wih,
                      const float* __restrict__ Whh,
                      const float* __restrict__ bih,
                      const float* __restrict__ bhh,
                      const float* __restrict__ Wlin,
                      const float* __restrict__ blin,
                      float* __restrict__ out)
{
    extern __shared__ char smem[];
    __bf16* sWlin  = (__bf16*)(smem + OFF_WLIN);   // [2][NAP][HID]
    __bf16* sH     = (__bf16*)(smem + OFF_H);      // [ROWS][HID]
    __bf16* sX     = (__bf16*)(smem + OFF_X);      // [2][ROWS][F_IN]
    float*  sGates = (float*)(smem + OFF_GATES);   // [ROWS][GATES]
    float*  sLog   = (float*)(smem + OFF_GATES);   // [ROWS][NAP] (post-loop alias)

    const int tid  = threadIdx.x;
    const int wave = tid >> 5;
    const int lane = tid & 31;
    const int n16  = lane & 15;   // N column in 16-wide tile (also A row M)
    const int hi   = lane >> 4;   // K-half selector
    const int row0 = blockIdx.x * ROWS;

    // ---- hoist loop-invariant weight B-fragments + bias into registers ----
    // Wave w owns gate columns [64w, 64w+64): 4 N-tiles. B-fragment layout:
    // lane column N=n16, lanes 0-15 K=0..15, lanes 16-31 K=16..31 (contig).
    v16bf wbX[4], wbH0[4], wbH1[4];
    float biasv[4];
    #pragma unroll
    for (int nt = 0; nt < 4; ++nt) {
        const int col = wave * 64 + nt * 16 + n16;
        const float* pih = Wih + (size_t)col * F_IN + hi * 16;
        const float* phh = Whh + (size_t)col * HID + hi * 16;
        #pragma unroll
        for (int e = 0; e < 16; ++e) {
            wbX[nt][e]  = (__bf16)pih[e];
            wbH0[nt][e] = (__bf16)phh[e];
            wbH1[nt][e] = (__bf16)phh[32 + e];
        }
        biasv[nt] = bih[col] + bhh[col];
    }

    // ---- init LDS state + stage x(0) ----
    for (int i = tid; i < ROWS * HID; i += NTHREADS) sH[i] = (__bf16)0.0f;
    for (int i = tid; i < (NAP - NA) * HID; i += NTHREADS) {
        sWlin[NA * HID + i]             = (__bf16)0.0f;   // pad rows, buffer 0
        sWlin[NAP * HID + NA * HID + i] = (__bf16)0.0f;   // pad rows, buffer 1
    }
    {
        const int r  = tid >> 3;
        const int f0 = (tid & 7) << 2;
        const float4 v = *(const float4*)(
            x + (((size_t)(row0 + r)) * T_STEPS + 0) * F_IN + f0);
        __bf16* d = sX + r * F_IN + f0;       // buffer 0
        d[0] = (__bf16)v.x; d[1] = (__bf16)v.y;
        d[2] = (__bf16)v.z; d[3] = (__bf16)v.w;
    }
    // Cell state: lane-stationary, lives in registers (thread owns (m,k0..k0+7)).
    const int em = tid >> 3;            // row 0..15
    const int ek = (tid & 7) << 3;      // k0: 0,8,...,56
    float creg[8];
    #pragma unroll
    for (int j = 0; j < 8; ++j) creg[j] = 0.0f;
    __syncthreads();

    // Persistent head (logits) accumulators: wave w owns asset tiles {w, w+4}.
    v8f accHead0 = {};
    v8f accHead1 = {};

    for (int t = 0; t < T_STEPS; ++t) {
        const int buf = t & 1;

        // ---- phase 1: gate WMMAs for t, head WMMAs for h_{t-1} ----
        v16bf aX, aH0, aH1;
        {
            const __bf16* px = sX + buf * (ROWS * F_IN) + n16 * F_IN;
            const __bf16* ph = sH + n16 * HID;
            #pragma unroll
            for (int e = 0; e < 16; ++e) {
                const int k = a_kmap(e, hi);
                aX[e]  = px[k];
                aH0[e] = ph[k];
                aH1[e] = ph[32 + k];
            }
        }
        #pragma unroll
        for (int nt = 0; nt < 4; ++nt) {
            v8f acc = {};
            acc = wmma_bf16(aX,  wbX[nt],  acc);
            acc = wmma_bf16(aH0, wbH0[nt], acc);
            acc = wmma_bf16(aH1, wbH1[nt], acc);
            const int col = wave * 64 + nt * 16 + n16;
            #pragma unroll
            for (int r = 0; r < 8; ++r)
                sGates[(r + 8 * hi) * GATES + col] = acc[r] + biasv[nt];
        }
        if (t > 0) {
            const __bf16* wl = sWlin + ((t - 1) & 1) * (NAP * HID);
            {
                const __bf16* p = wl + (wave * 16 + n16) * HID + hi * 16;
                v16bf b0, b1;
                #pragma unroll
                for (int e = 0; e < 16; ++e) { b0[e] = p[e]; b1[e] = p[32 + e]; }
                accHead0 = wmma_bf16(aH0, b0, accHead0);
                accHead0 = wmma_bf16(aH1, b1, accHead0);
            }
            if (wave < 3) {
                const __bf16* p = wl + ((wave + 4) * 16 + n16) * HID + hi * 16;
                v16bf b0, b1;
                #pragma unroll
                for (int e = 0; e < 16; ++e) { b0[e] = p[e]; b1[e] = p[32 + e]; }
                accHead1 = wmma_bf16(aH0, b0, accHead1);
                accHead1 = wmma_bf16(aH1, b1, accHead1);
            }
        }
        __syncthreads();

        // ---- phase 2: elementwise cell update + stage x(t+1), W_lin(t) ----
        #pragma unroll
        for (int j = 0; j < 8; ++j) {
            const int k = ek + j;
            const float gi = sGates[em * GATES + k];
            const float gf = sGates[em * GATES + 64 + k];
            const float gg = sGates[em * GATES + 128 + k];
            const float go = sGates[em * GATES + 192 + k];
            const float c  = sigmoid_f(gf) * creg[j] + sigmoid_f(gi) * tanh_f(gg);
            creg[j] = c;
            sH[em * HID + k] = (__bf16)(sigmoid_f(go) * tanh_f(c));
        }
        if (t + 1 < T_STEPS) {
            const int r  = tid >> 3;
            const int f0 = (tid & 7) << 2;
            const float4 v = *(const float4*)(
                x + (((size_t)(row0 + r)) * T_STEPS + (t + 1)) * F_IN + f0);
            __bf16* d = sX + ((t + 1) & 1) * (ROWS * F_IN) + r * F_IN + f0;
            d[0] = (__bf16)v.x; d[1] = (__bf16)v.y;
            d[2] = (__bf16)v.z; d[3] = (__bf16)v.w;
        }
        {   // W_lin slice t -> buffer (t&1); consumed by phase 1 of t+1.
            const float* wl = Wlin + (size_t)t * HID;
            __bf16* dst = sWlin + buf * (NAP * HID);
            for (int i = tid; i < NA * (HID / 4); i += NTHREADS) {
                const int a  = i >> 4;
                const int kq = (i & 15) << 2;
                const float4 v = *(const float4*)(wl + (size_t)a * (T_STEPS * HID) + kq);
                __bf16* d = dst + a * HID + kq;
                d[0] = (__bf16)v.x; d[1] = (__bf16)v.y;
                d[2] = (__bf16)v.z; d[3] = (__bf16)v.w;
            }
        }
        if (t + 1 < T_STEPS && tid < NA) {  // warm L2 (global_prefetch_b8)
            __builtin_prefetch(Wlin + (size_t)tid * (T_STEPS * HID) + (size_t)(t + 1) * HID, 0, 1);
        }
        __syncthreads();
    }

    // ---- epilogue: head contribution of h_{T-1} (slice already staged) ----
    {
        v16bf aH0, aH1;
        const __bf16* ph = sH + n16 * HID;
        #pragma unroll
        for (int e = 0; e < 16; ++e) {
            const int k = a_kmap(e, hi);
            aH0[e] = ph[k];
            aH1[e] = ph[32 + k];
        }
        const __bf16* wl = sWlin + ((T_STEPS - 1) & 1) * (NAP * HID);
        {
            const __bf16* p = wl + (wave * 16 + n16) * HID + hi * 16;
            v16bf b0, b1;
            #pragma unroll
            for (int e = 0; e < 16; ++e) { b0[e] = p[e]; b1[e] = p[32 + e]; }
            accHead0 = wmma_bf16(aH0, b0, accHead0);
            accHead0 = wmma_bf16(aH1, b1, accHead0);
        }
        if (wave < 3) {
            const __bf16* p = wl + ((wave + 4) * 16 + n16) * HID + hi * 16;
            v16bf b0, b1;
            #pragma unroll
            for (int e = 0; e < 16; ++e) { b0[e] = p[e]; b1[e] = p[32 + e]; }
            accHead1 = wmma_bf16(aH0, b0, accHead1);
            accHead1 = wmma_bf16(aH1, b1, accHead1);
        }
    }

    // Spill logits (C/D layout: lane<16 -> M=r, N=lane; lane>=16 -> M=r+8).
    #pragma unroll
    for (int r = 0; r < 8; ++r)
        sLog[(r + 8 * hi) * NAP + wave * 16 + n16] = accHead0[r];
    if (wave < 3) {
        #pragma unroll
        for (int r = 0; r < 8; ++r)
            sLog[(r + 8 * hi) * NAP + (wave + 4) * 16 + n16] = accHead1[r];
    }
    __syncthreads();

    // ---- bias + softmax, one batch row per thread (threads 0..15) ----
    if (tid < ROWS) {
        float* lr = sLog + tid * NAP;
        float mx = -3.4e38f;
        for (int a = 0; a < NA; ++a) {
            const float v = lr[a] + blin[a];
            lr[a] = v;
            mx = fmaxf(mx, v);
        }
        float s = 0.0f;
        for (int a = 0; a < NA; ++a) {
            const float e = __expf(lr[a] - mx);
            lr[a] = e;
            s += e;
        }
        const float inv = 1.0f / s;
        float* orow = out + (size_t)(row0 + tid) * NA;
        for (int a = 0; a < NA; ++a) orow[a] = lr[a] * inv;
    }
}

extern "C" void kernel_launch(void* const* d_in, const int* in_sizes, int n_in,
                              void* d_out, int out_size, void* d_ws, size_t ws_size,
                              hipStream_t stream) {
    const float* x    = (const float*)d_in[0];
    const float* Wih  = (const float*)d_in[1];
    const float* Whh  = (const float*)d_in[2];
    const float* bih  = (const float*)d_in[3];
    const float* bhh  = (const float*)d_in[4];
    const float* Wlin = (const float*)d_in[5];
    const float* blin = (const float*)d_in[6];
    float* out = (float*)d_out;

    const int B    = in_sizes[0] / (T_STEPS * F_IN);  // 8192
    const int grid = B / ROWS;                        // 512 workgroups
    lstm_head_kernel<<<grid, NTHREADS, SMEM_BYTES, stream>>>(
        x, Wih, Whh, bih, bhh, Wlin, blin, out);
}
